// DeformBlock_63617055588616
// MI455X (gfx1250) — matrix operand order repaired and checked by reading, not compiled, not run on hardware
//
#include <hip/hip_runtime.h>
#include <hip/hip_bf16.h>

#define HH 128
#define WWID 128
#define BIMGS 4
#define CCH 64
#define HWSZ (HH*WWID)        // 16384
#define NPIX (BIMGS*HWSZ)     // 65536
#define KDIM 576              // 9*64
#define NKI 18                // KDIM/32
#define OFFPITCH 28
#define BN_EPS 1e-5f

typedef __attribute__((ext_vector_type(16))) _Float16 v16h;
typedef __attribute__((ext_vector_type(8)))  _Float16 h8;
typedef __attribute__((ext_vector_type(8)))  float    v8f;
typedef __attribute__((ext_vector_type(4)))  float    f4;

#if defined(__has_builtin)
#if __has_builtin(__builtin_amdgcn_global_load_async_to_lds_b128) && \
    __has_builtin(__builtin_amdgcn_s_wait_asynccnt)
#define USE_ASYNC_LDS 1
#endif
#endif

#ifdef USE_ASYNC_LDS
// Builtin signature (from hipcc diagnostic): param0 = int4(vector_size 16) in
// AS1 ("__device__"), param1 = int4 in AS3, then imm offset + imm cpol.
typedef int vi4 __attribute__((vector_size(16)));
typedef __attribute__((address_space(1))) vi4 gvi4_t;
typedef __attribute__((address_space(3))) vi4 lvi4_t;
#endif

__device__ __forceinline__ int iclamp(int v, int lo, int hi) {
    return v < lo ? lo : (v > hi ? hi : v);
}

// Stage `bytes` (multiple of 128*16) from global to LDS, cooperatively by a
// 128-thread block.  Uses the CDNA5 async global->LDS path when available
// (ASYNCcnt-tracked, overlaps with subsequent VALU/VMEM work); otherwise a
// synchronous uint4 copy.
__device__ __forceinline__ void stage_to_lds(const _Float16* __restrict__ g,
                                             _Float16* l, int bytes, int tid) {
#ifdef USE_ASYNC_LDS
    for (int i = tid * 16; i < bytes; i += 128 * 16) {
        gvi4_t* gp = (gvi4_t*)(uintptr_t)((const char*)g + i);
        lvi4_t* lp = (lvi4_t*)(unsigned)(uintptr_t)((char*)l + i);
        __builtin_amdgcn_global_load_async_to_lds_b128(gp, lp, 0, 0);
    }
#else
    const uint4* s = (const uint4*)g;
    uint4* d = (uint4*)l;
    for (int i = tid; i < bytes / 16; i += 128) d[i] = s[i];
#endif
}

__device__ __forceinline__ void stage_wait() {
#ifdef USE_ASYNC_LDS
    __builtin_amdgcn_s_wait_asynccnt(0);
#endif
}

// ---------------------------------------------------------------------------
// Pre-pack OIHW f32 weights into f16 WMMA B-fragment order.
// K index = tap*64 + c.  B-frag mapping (16x16x32 f16, wave32):
//   lane L holds column n = t*16 + (L&15); halves h cover k = ki*32 + (L<16?0:8) + (h<8?0:8) + h
// packed[( (ki*NT + t)*32 + L )*16 + h]
// ---------------------------------------------------------------------------
__global__ __launch_bounds__(256) void prep_weights(const float* __restrict__ w,
                                                    int CO, int NT,
                                                    _Float16* __restrict__ dst) {
    int idx = blockIdx.x * 256 + threadIdx.x;
    int total = NKI * NT * 512;
    if (idx >= total) return;
    int h  = idx & 15;
    int L  = (idx >> 4) & 31;
    int tm = idx >> 9;
    int t  = tm % NT;
    int ki = tm / NT;
    int k  = ki * 32 + ((L >= 16) ? 8 : 0) + ((h >= 8) ? 8 : 0) + h;
    int n  = t * 16 + (L & 15);
    int c  = k & 63;
    int tap = k >> 6;
    float val = (n < CO) ? w[(n * 64 + c) * 9 + tap] : 0.0f;
    dst[idx] = (_Float16)val;
}

// ---------------------------------------------------------------------------
// NCHW -> NHWC with LDS tile transpose (both directions coalesced)
// ---------------------------------------------------------------------------
__global__ __launch_bounds__(256) void nchw_to_nhwc(const float* __restrict__ src,
                                                    float* __restrict__ dst) {
    __shared__ float tile[64][65];
    int p0   = blockIdx.x * 64;
    int bimg = p0 >> 14;
    int prem = p0 & (HWSZ - 1);
    int tid  = threadIdx.x;
#pragma unroll
    for (int k = 0; k < 16; ++k) {
        int o = tid + k * 256;
        int i = o & 63, c = o >> 6;
        tile[i][c] = src[(size_t)(bimg * CCH + c) * HWSZ + prem + i];
    }
    __syncthreads();
#pragma unroll
    for (int k = 0; k < 16; ++k) {
        int o = tid + k * 256;
        int c = o & 63, i = o >> 6;
        dst[(size_t)(p0 + i) * CCH + c] = tile[i][c];
    }
}

// ---------------------------------------------------------------------------
// Offset conv: 3x3 SAME conv 64 -> 27 via WMMA (N padded to 32).
// Writes offb[pix*28 + slot]: slot k = dy_k, 9+k = dx_k, 18+k = sigmoid(mask_k)
// dyn LDS: B 18432 halves (36864B) + A 4*9216 halves (73728B)
// ---------------------------------------------------------------------------
__global__ __launch_bounds__(128) void offset_conv_wmma(const float* __restrict__ in,
                                                        const _Float16* __restrict__ wB,
                                                        const float* __restrict__ bias,
                                                        float* __restrict__ offb) {
    extern __shared__ char smem[];
    _Float16* blds  = (_Float16*)smem;                 // 18432 halves
    _Float16* atile = (_Float16*)(smem + 18432 * 2);   // 4 * 9216 halves
    int tid = threadIdx.x, wave = tid >> 5, lane = tid & 31;

    // fire off B staging (async path overlaps with the A-tile build below)
    stage_to_lds(wB, blds, 18432 * 2, tid);

    // fill A-tile in fragment layout: lane = (pixel p, channel-half)
    int p = lane & 15, chalf = lane >> 4;
    int pix = blockIdx.x * 64 + wave * 16 + p;
    int bimg = pix >> 14, rem = pix & (HWSZ - 1);
    int y = rem >> 7, x = rem & (WWID - 1);
    _Float16* aw = atile + wave * 9216;
#pragma unroll
    for (int tap = 0; tap < 9; ++tap) {
        int yy = y + tap / 3 - 1, xx = x + tap % 3 - 1;
        bool valid = (yy >= 0 && yy < HH && xx >= 0 && xx < WWID);
        const float* src = in + ((size_t)(bimg * HH + (valid ? yy : 0)) * WWID +
                                 (valid ? xx : 0)) * CCH + chalf * 32;
        int ki = tap * 2 + chalf;
#pragma unroll
        for (int blk = 0; blk < 4; ++blk) {
            int jj = blk * 8;
            f4 a0 = valid ? *(const f4*)(src + jj)     : (f4){0.f, 0.f, 0.f, 0.f};
            f4 a1 = valid ? *(const f4*)(src + jj + 4) : (f4){0.f, 0.f, 0.f, 0.f};
            h8 hv;
#pragma unroll
            for (int u = 0; u < 4; ++u) { hv[u] = (_Float16)a0[u]; hv[u + 4] = (_Float16)a1[u]; }
            int dl = p + ((blk & 1) << 4);
            int h0 = (blk & 2) << 2;
            *(h8*)(aw + ((ki * 32 + dl) << 4) + h0) = hv;
        }
    }
    stage_wait();
    __syncthreads();

    // K-loop: 18 x 2 WMMAs
    int n = lane & 15;
    v8f acc0, acc1;
#pragma unroll
    for (int u = 0; u < 8; ++u) { acc0[u] = 0.f; acc1[u] = 0.f; }
#pragma unroll
    for (int ki = 0; ki < NKI; ++ki) {
        v16h afr = *(const v16h*)(aw + ((ki * 32 + lane) << 4));
        v16h bf0 = *(const v16h*)(blds + (((ki * 2 + 0) * 32 + lane) << 4));
        v16h bf1 = *(const v16h*)(blds + (((ki * 2 + 1) * 32 + lane) << 4));
        acc0 = __builtin_amdgcn_wmma_f32_16x16x32_f16(false, afr, false, bf0,
                                                      (short)0, acc0, false, false);
        acc1 = __builtin_amdgcn_wmma_f32_16x16x32_f16(false, afr, false, bf1,
                                                      (short)0, acc1, false, false);
    }

    // epilogue: split into dy / dx / sigmoid(mask)
#pragma unroll
    for (int t = 0; t < 2; ++t) {
#pragma unroll
        for (int r = 0; r < 8; ++r) {
            int o = t * 16 + n;
            if (o >= 27) continue;
            float v = (t ? acc1[r] : acc0[r]) + bias[o];
            int m = ((lane >> 4) << 3) + r;
            int pg = blockIdx.x * 64 + wave * 16 + m;
            int slot;
            if (o < 18) {
                slot = (o & 1) ? 9 + (o >> 1) : (o >> 1);
            } else {
                v = 1.0f / (1.0f + expf(-v));
                slot = o;
            }
            offb[(size_t)pg * OFFPITCH + slot] = v;
        }
    }
}

// ---------------------------------------------------------------------------
// Deformable conv: bilinear-gather -> mask-modulate -> WMMA GEMM [16,576]x[576,64]
// plus per-workgroup BN partial sums (deterministic, no atomics).
// dyn LDS: B 73728B + A 73728B + 2KB reduce
// ---------------------------------------------------------------------------
__global__ __launch_bounds__(128) void deform_wmma(const float* __restrict__ in,
                                                   const float* __restrict__ offb,
                                                   const _Float16* __restrict__ wB,
                                                   const float* __restrict__ bias,
                                                   float* __restrict__ ybuf,
                                                   float* __restrict__ psum,
                                                   float* __restrict__ psumsq) {
    extern __shared__ char smem[];
    _Float16* blds  = (_Float16*)smem;                 // 36864 halves
    _Float16* atile = (_Float16*)(smem + 36864 * 2);   // 36864 halves
    float* red0 = (float*)(smem + 36864 * 4);          // [4][64]
    float* red1 = red0 + 256;
    int tid = threadIdx.x, wave = tid >> 5, lane = tid & 31;

    // fire off B staging (async path overlaps with the gather phase below)
    stage_to_lds(wB, blds, 36864 * 2, tid);

    int p = lane & 15, chalf = lane >> 4;
    int pix = blockIdx.x * 64 + wave * 16 + p;
    int bimg = pix >> 14, rem = pix & (HWSZ - 1);
    int y = rem >> 7, x = rem & (WWID - 1);
    const float* op = offb + (size_t)pix * OFFPITCH;
    _Float16* aw = atile + wave * 9216;

#pragma unroll
    for (int tap = 0; tap < 9; ++tap) {
        float dy = op[tap], dx = op[9 + tap], mk = op[18 + tap];
        float py = (float)(y + tap / 3 - 1) + dy;
        float px = (float)(x + tap % 3 - 1) + dx;
        float fy = floorf(py), fx = floorf(px);
        int y0 = (int)fy, x0 = (int)fx;
        int y1 = y0 + 1, x1 = x0 + 1;
        float wy1 = py - fy, wx1 = px - fx;
        float wy0 = 1.f - wy1, wx0 = 1.f - wx1;
        float vy0 = (y0 >= 0 && y0 < HH)   ? 1.f : 0.f;
        float vy1 = (y1 >= 0 && y1 < HH)   ? 1.f : 0.f;
        float vx0 = (x0 >= 0 && x0 < WWID) ? 1.f : 0.f;
        float vx1 = (x1 >= 0 && x1 < WWID) ? 1.f : 0.f;
        float w00 = wy0 * wx0 * vy0 * vx0;
        float w01 = wy0 * wx1 * vy0 * vx1;
        float w10 = wy1 * wx0 * vy1 * vx0;
        float w11 = wy1 * wx1 * vy1 * vx1;
        int y0c = iclamp(y0, 0, HH - 1),   x0c = iclamp(x0, 0, WWID - 1);
        int y1c = iclamp(y1, 0, HH - 1),   x1c = iclamp(x1, 0, WWID - 1);
        const float* p00 = in + ((size_t)(bimg * HH + y0c) * WWID + x0c) * CCH + chalf * 32;
        const float* p01 = in + ((size_t)(bimg * HH + y0c) * WWID + x1c) * CCH + chalf * 32;
        const float* p10 = in + ((size_t)(bimg * HH + y1c) * WWID + x0c) * CCH + chalf * 32;
        const float* p11 = in + ((size_t)(bimg * HH + y1c) * WWID + x1c) * CCH + chalf * 32;
        int ki = tap * 2 + chalf;
#pragma unroll
        for (int blk = 0; blk < 4; ++blk) {
            int jj = blk * 8;
            f4 g00a = *(const f4*)(p00 + jj), g00b = *(const f4*)(p00 + jj + 4);
            f4 g01a = *(const f4*)(p01 + jj), g01b = *(const f4*)(p01 + jj + 4);
            f4 g10a = *(const f4*)(p10 + jj), g10b = *(const f4*)(p10 + jj + 4);
            f4 g11a = *(const f4*)(p11 + jj), g11b = *(const f4*)(p11 + jj + 4);
            h8 hv;
#pragma unroll
            for (int u = 0; u < 4; ++u) {
                float va = w00 * g00a[u] + w01 * g01a[u] + w10 * g10a[u] + w11 * g11a[u];
                float vb = w00 * g00b[u] + w01 * g01b[u] + w10 * g10b[u] + w11 * g11b[u];
                hv[u]     = (_Float16)(va * mk);
                hv[u + 4] = (_Float16)(vb * mk);
            }
            int dl = p + ((blk & 1) << 4);
            int h0 = (blk & 2) << 2;
            *(h8*)(aw + ((ki * 32 + dl) << 4) + h0) = hv;
        }
    }
    stage_wait();
    __syncthreads();

    int n = lane & 15;
    v8f acc[4];
#pragma unroll
    for (int t = 0; t < 4; ++t) {
        float bv = bias[t * 16 + n];
#pragma unroll
        for (int u = 0; u < 8; ++u) acc[t][u] = bv;
    }
#pragma unroll
    for (int ki = 0; ki < NKI; ++ki) {
        v16h afr = *(const v16h*)(aw + ((ki * 32 + lane) << 4));
#pragma unroll
        for (int t = 0; t < 4; ++t) {
            v16h bf = *(const v16h*)(blds + (((ki * 4 + t) * 32 + lane) << 4));
            acc[t] = __builtin_amdgcn_wmma_f32_16x16x32_f16(false, afr, false, bf,
                                                            (short)0, acc[t], false, false);
        }
    }

    // store y (NHWC) + per-channel partial sums for BN
#pragma unroll
    for (int t = 0; t < 4; ++t) {
        float sv = 0.f, qv = 0.f;
#pragma unroll
        for (int r = 0; r < 8; ++r) {
            float v = acc[t][r];
            int m  = ((lane >> 4) << 3) + r;
            int pg = blockIdx.x * 64 + wave * 16 + m;
            ybuf[(size_t)pg * CCH + t * 16 + n] = v;
            sv += v;
            qv += v * v;
        }
        sv += __shfl_xor(sv, 16);
        qv += __shfl_xor(qv, 16);
        if (lane < 16) {
            red0[wave * 64 + t * 16 + n] = sv;
            red1[wave * 64 + t * 16 + n] = qv;
        }
    }
    __syncthreads();
    if (tid < 64) {
        float S = 0.f, Q = 0.f;
#pragma unroll
        for (int w = 0; w < 4; ++w) { S += red0[w * 64 + tid]; Q += red1[w * 64 + tid]; }
        psum[(size_t)blockIdx.x * 64 + tid]   = S;
        psumsq[(size_t)blockIdx.x * 64 + tid] = Q;
    }
}

// ---------------------------------------------------------------------------
// Fold 1024 per-WG partials into per-channel scale/shift
// ---------------------------------------------------------------------------
__global__ __launch_bounds__(256) void reduce_stats(const float* __restrict__ psum,
                                                    const float* __restrict__ psumsq,
                                                    const float* __restrict__ gamma,
                                                    const float* __restrict__ beta,
                                                    float* __restrict__ stats) {
    __shared__ float rs[256], rq[256];
    int c = blockIdx.x, tid = threadIdx.x;
    float s = 0.f, q = 0.f;
    for (int w = tid; w < 1024; w += 256) {
        s += psum[(size_t)w * 64 + c];
        q += psumsq[(size_t)w * 64 + c];
    }
    rs[tid] = s; rq[tid] = q;
    __syncthreads();
    for (int st = 128; st > 0; st >>= 1) {
        if (tid < st) { rs[tid] += rs[tid + st]; rq[tid] += rq[tid + st]; }
        __syncthreads();
    }
    if (tid == 0) {
        float mean = rs[0] / (float)NPIX;
        float var  = rq[0] / (float)NPIX - mean * mean;
        float sc   = gamma[c] * rsqrtf(var + BN_EPS);
        stats[2 * c]     = sc;
        stats[2 * c + 1] = beta[c] - mean * sc;
    }
}

__global__ __launch_bounds__(256) void bn_relu_nhwc(const float* __restrict__ yb,
                                                    const float* __restrict__ stats,
                                                    float* __restrict__ out) {
    int i = blockIdx.x * 256 + threadIdx.x;
    int c = i & 63;
    float v = yb[i] * stats[2 * c] + stats[2 * c + 1];
    out[i] = fmaxf(v, 0.f);
}

// final stage: BN+ReLU and NHWC -> NCHW transpose to d_out
__global__ __launch_bounds__(256) void bn_relu_nchw(const float* __restrict__ yb,
                                                    const float* __restrict__ stats,
                                                    float* __restrict__ out) {
    __shared__ float tile[64][65];
    int p0 = blockIdx.x * 64;
    int bimg = p0 >> 14;
    int prem = p0 & (HWSZ - 1);
    int tid = threadIdx.x;
#pragma unroll
    for (int k = 0; k < 16; ++k) {
        int o = tid + k * 256;
        int c = o & 63, i = o >> 6;
        float v = yb[(size_t)(p0 + i) * CCH + c] * stats[2 * c] + stats[2 * c + 1];
        tile[i][c] = fmaxf(v, 0.f);
    }
    __syncthreads();
#pragma unroll
    for (int k = 0; k < 16; ++k) {
        int o = tid + k * 256;
        int i = o & 63, c = o >> 6;
        out[(size_t)(bimg * CCH + c) * HWSZ + prem + i] = tile[i][c];
    }
}

extern "C" void kernel_launch(void* const* d_in, const int* in_sizes, int n_in,
                              void* d_out, int out_size, void* d_ws, size_t ws_size,
                              hipStream_t stream) {
    (void)in_sizes; (void)n_in; (void)out_size; (void)ws_size;
    const float* x      = (const float*)d_in[0];
    const float* w_off1 = (const float*)d_in[1];
    const float* b_off1 = (const float*)d_in[2];
    const float* w_def1 = (const float*)d_in[3];
    const float* b_def1 = (const float*)d_in[4];
    const float* gamma1 = (const float*)d_in[5];
    const float* beta1  = (const float*)d_in[6];
    const float* w_off2 = (const float*)d_in[7];
    const float* b_off2 = (const float*)d_in[8];
    const float* w_def2 = (const float*)d_in[9];
    const float* b_def2 = (const float*)d_in[10];
    const float* gamma2 = (const float*)d_in[11];
    const float* beta2  = (const float*)d_in[12];

    char* ws = (char*)d_ws;
    size_t o = 0;
    float* bufA = (float*)(ws + o); o += (size_t)NPIX * CCH * 4;       // x NHWC
    float* bufB = (float*)(ws + o); o += (size_t)NPIX * CCH * 4;       // h NHWC
    float* bufC = (float*)(ws + o); o += (size_t)NPIX * CCH * 4;       // y NHWC
    float* offb = (float*)(ws + o); o += (size_t)NPIX * OFFPITCH * 4;  // dy/dx/mask
    _Float16* wd1 = (_Float16*)(ws + o); o += 36864 * 2;
    _Float16* wo1 = (_Float16*)(ws + o); o += 18432 * 2;
    _Float16* wd2 = (_Float16*)(ws + o); o += 36864 * 2;
    _Float16* wo2 = (_Float16*)(ws + o); o += 18432 * 2;
    float* psum = (float*)(ws + o); o += 1024 * 64 * 4;
    float* psq  = (float*)(ws + o); o += 1024 * 64 * 4;
    float* stats = (float*)(ws + o); o += 128 * 4;

    prep_weights<<<(36864 + 255) / 256, 256, 0, stream>>>(w_def1, 64, 4, wd1);
    prep_weights<<<(18432 + 255) / 256, 256, 0, stream>>>(w_off1, 27, 2, wo1);
    prep_weights<<<(36864 + 255) / 256, 256, 0, stream>>>(w_def2, 64, 4, wd2);
    prep_weights<<<(18432 + 255) / 256, 256, 0, stream>>>(w_off2, 27, 2, wo2);
    nchw_to_nhwc<<<1024, 256, 0, stream>>>(x, bufA);

    const size_t lds_off = 18432 * 2 + 4 * 9216 * 2;            // 110592 B
    const size_t lds_def = 36864 * 2 + 36864 * 2 + 2 * 256 * 4; // 149504 B

    // stage 1
    offset_conv_wmma<<<1024, 128, lds_off, stream>>>(bufA, wo1, b_off1, offb);
    deform_wmma<<<1024, 128, lds_def, stream>>>(bufA, offb, wd1, b_def1, bufC, psum, psq);
    reduce_stats<<<64, 256, 0, stream>>>(psum, psq, gamma1, beta1, stats);
    bn_relu_nhwc<<<(NPIX * CCH) / 256, 256, 0, stream>>>(bufC, stats, bufB);

    // stage 2
    offset_conv_wmma<<<1024, 128, lds_off, stream>>>(bufB, wo2, b_off2, offb);
    deform_wmma<<<1024, 128, lds_def, stream>>>(bufB, offb, wd2, b_def2, bufC, psum, psq);
    reduce_stats<<<64, 256, 0, stream>>>(psum, psq, gamma2, beta2, stats);
    bn_relu_nchw<<<1024, 256, 0, stream>>>(bufC, stats, (float*)d_out);
}
